// MoELayer_8323646619996
// MI455X (gfx1250) — compile-verified, matrix operand order checked by reading, not run on hardware
//
#include <hip/hip_runtime.h>
#include <hip/hip_bf16.h>

// ---------------- types ----------------
typedef __attribute__((ext_vector_type(16))) _Float16 v16h;
typedef __attribute__((ext_vector_type(8)))  _Float16 v8h;
typedef __attribute__((ext_vector_type(4)))  _Float16 v4h;
typedef __attribute__((ext_vector_type(8)))  float    v8f;
typedef __attribute__((ext_vector_type(4)))  float    v4f;

typedef __attribute__((address_space(3))) _Float16 lds_h;

union FragAB { v16h v; v8h h[2]; };

__device__ inline v8f v8f_zero() {
  v8f v;
#pragma unroll
  for (int i = 0; i < 8; ++i) v[i] = 0.f;
  return v;
}
__device__ inline v8h v8h_zero() {
  v8h v;
#pragma unroll
  for (int i = 0; i < 8; ++i) v[i] = (_Float16)0.f;
  return v;
}
__device__ inline v4f v4f_zero() {
  v4f v; v[0] = 0.f; v[1] = 0.f; v[2] = 0.f; v[3] = 0.f;
  return v;
}

// async copy: 16B global -> LDS, tracked by ASYNCcnt (CDNA5)
__device__ inline void async_load_lds_b128(unsigned lds_off, const void* gaddr) {
  asm volatile("global_load_async_to_lds_b128 %0, %1, off"
               :: "v"(lds_off), "v"(gaddr) : "memory");
}
__device__ inline void wait_asynccnt0() {
  asm volatile("s_wait_asynccnt 0x0" ::: "memory");
}

// ---------------- problem dims ----------------
#define T_TOK   8192      // B*S
#define DDIM    1024
#define HDIM    2048
#define NEXP    8
#define TOPK    2
#define NASSIGN (T_TOK * TOPK)   // 16384

// ---------------- GEMM tiling ----------------
#define BM   128
#define BN   64
#define BK   32
#define LDK  40   // padded LDS K-stride in halfs (breaks bank conflicts)

// =====================================================================
// Kernel 0: zero the output (atomic accumulation target)
// =====================================================================
__global__ void moe_zero_out(float* __restrict__ out, int n4) {
  int i = blockIdx.x * blockDim.x + threadIdx.x;
  if (i < n4) ((v4f*)out)[i] = v4f_zero();
}

// =====================================================================
// Kernel 1: gating — one wave per token; top-2 + softmax over top-2
// =====================================================================
__global__ __launch_bounds__(256)
void moe_gate_topk(const float* __restrict__ x, const float* __restrict__ Wg,
                   int* __restrict__ top_idx, float* __restrict__ top_prob) {
  const int wave  = threadIdx.x >> 5;
  const int lane  = threadIdx.x & 31;
  const int token = blockIdx.x * 8 + wave;
  const float* xr = x + (size_t)token * DDIM;

  float acc[NEXP];
#pragma unroll
  for (int e = 0; e < NEXP; ++e) acc[e] = 0.f;

  for (int d = lane; d < DDIM; d += 32) {
    float xv = xr[d];
#pragma unroll
    for (int e = 0; e < NEXP; ++e) acc[e] += xv * Wg[e * DDIM + d];
  }
#pragma unroll
  for (int e = 0; e < NEXP; ++e) {
#pragma unroll
    for (int off = 16; off > 0; off >>= 1)
      acc[e] += __shfl_xor(acc[e], off, 32);
  }
  if (lane == 0) {
    float v1 = -__builtin_inff(), v2 = -__builtin_inff();
    int i1 = 0, i2 = 0;
#pragma unroll
    for (int e = 0; e < NEXP; ++e) {
      float s = acc[e];
      if (s > v1)      { v2 = v1; i2 = i1; v1 = s; i1 = e; }
      else if (s > v2) { v2 = s;  i2 = e; }
    }
    float p1 = 1.f / (1.f + __expf(v2 - v1));
    top_idx [token * 2 + 0] = i1;
    top_idx [token * 2 + 1] = i2;
    top_prob[token * 2 + 0] = p1;
    top_prob[token * 2 + 1] = 1.f - p1;
  }
}

// =====================================================================
// Kernel 2: deterministic per-expert assignment lists (token order)
// =====================================================================
__global__ void moe_build_lists(const int* __restrict__ top_idx,
                                const float* __restrict__ top_prob,
                                int* __restrict__ counts, int* __restrict__ offsets,
                                int* __restrict__ assign_token,
                                float* __restrict__ assign_prob) {
  __shared__ int sCnt[NEXP];
  __shared__ int sOff[NEXP];
  const int e = threadIdx.x;
  if (e < NEXP) {
    int c = 0;
    for (int t = 0; t < T_TOK; ++t)
#pragma unroll
      for (int k = 0; k < TOPK; ++k)
        if (top_idx[t * TOPK + k] == e) ++c;
    sCnt[e] = c; counts[e] = c;
  }
  __syncthreads();
  if (threadIdx.x == 0) {
    int run = 0;
    for (int i = 0; i < NEXP; ++i) { sOff[i] = run; offsets[i] = run; run += sCnt[i]; }
  }
  __syncthreads();
  if (e < NEXP) {
    int pos = sOff[e];
    for (int t = 0; t < T_TOK; ++t)
#pragma unroll
      for (int k = 0; k < TOPK; ++k)
        if (top_idx[t * TOPK + k] == e) {
          assign_token[pos] = t;
          assign_prob [pos] = top_prob[t * TOPK + k];
          ++pos;
        }
  }
}

// =====================================================================
// Kernel 3: fused SwiGLU stage — hidden = silu(x@W1e^T) * (x@W2e^T)
// Block tile 128x64, K-step 32, f16 WMMA w/ f32 accumulate.
// Gather pointers hoisted out of K loop; register-prefetch pipeline.
// =====================================================================
__global__ __launch_bounds__(256)
void moe_expert_ffn1(const float* __restrict__ x,
                     const float* __restrict__ W1, const float* __restrict__ W2,
                     const int* __restrict__ counts, const int* __restrict__ offsets,
                     const int* __restrict__ assign_token,
                     _Float16* __restrict__ hidden) {
  const int e = blockIdx.z;
  const int count = counts[e];
  const int rowBase = blockIdx.y * BM;
  if (rowBase >= count) return;               // whole block exits: EXEC stays full
  const int hBase = blockIdx.x * BN;
  const int aOff  = offsets[e];

  __shared__ _Float16 sA [BM * LDK];
  __shared__ _Float16 sB1[BN * LDK];
  __shared__ _Float16 sB2[BN * LDK];

  const int tid   = threadIdx.x;
  const int lane  = tid & 31;
  const int wave  = tid >> 5;
  const int waveM = wave >> 1;   // 0..3  (32 rows each)
  const int waveN = wave & 1;    // 0..1  (32 cols each)
  const int mRow  = lane & 15;
  const int khalf = lane >> 4;   // ISA 16-bit A/B layout: K chunks {kh*8, 16+kh*8}

  const float* W1e = W1 + (size_t)e * HDIM * DDIM;
  const float* W2e = W2 + (size_t)e * HDIM * DDIM;

  // ---- loop-invariant gather: per-thread source pointers & LDS slots ----
  const float* aSrc[4]; int aLds[4];
#pragma unroll
  for (int it = 0; it < 4; ++it) {
    int slot = tid + it * 256;               // 1024 float4-slots (128x32)
    int r = slot >> 3, c4 = slot & 7;
    int gr = rowBase + r;
    aSrc[it] = nullptr;
    if (gr < count) {
      int tok = assign_token[aOff + gr];
      aSrc[it] = x + (size_t)tok * DDIM + c4 * 4;
    }
    aLds[it] = r * LDK + c4 * 4;
  }
  const float* bSrc1[2]; const float* bSrc2[2]; int bLds[2];
#pragma unroll
  for (int it = 0; it < 2; ++it) {
    int slot = tid + it * 256;               // 512 float4-slots (64x32)
    int r = slot >> 3, c4 = slot & 7;
    bSrc1[it] = W1e + (size_t)(hBase + r) * DDIM + c4 * 4;
    bSrc2[it] = W2e + (size_t)(hBase + r) * DDIM + c4 * 4;
    bLds[it]  = r * LDK + c4 * 4;
  }

  v4f pA[4], pB1[2], pB2[2];
  auto fetch = [&](int k0) {
#pragma unroll
    for (int it = 0; it < 4; ++it)
      pA[it] = aSrc[it] ? *(const v4f*)(aSrc[it] + k0) : v4f_zero();
#pragma unroll
    for (int it = 0; it < 2; ++it) {
      pB1[it] = *(const v4f*)(bSrc1[it] + k0);
      pB2[it] = *(const v4f*)(bSrc2[it] + k0);
    }
  };

  v8f accG[2][2], accU[2][2];
#pragma unroll
  for (int i = 0; i < 2; ++i)
#pragma unroll
    for (int j = 0; j < 2; ++j) { accG[i][j] = v8f_zero(); accU[i][j] = v8f_zero(); }

  fetch(0);
  for (int k0 = 0; k0 < DDIM; k0 += BK) {
    __syncthreads();                         // LDS free from previous iteration
    // ---- convert & stage prefetched registers to LDS ----
#pragma unroll
    for (int it = 0; it < 4; ++it) {
      v4h h;
#pragma unroll
      for (int q = 0; q < 4; ++q) h[q] = (_Float16)pA[it][q];
      *(v4h*)&sA[aLds[it]] = h;
    }
#pragma unroll
    for (int it = 0; it < 2; ++it) {
      v4h h1, h2;
#pragma unroll
      for (int q = 0; q < 4; ++q) { h1[q] = (_Float16)pB1[it][q]; h2[q] = (_Float16)pB2[it][q]; }
      *(v4h*)&sB1[bLds[it]] = h1;
      *(v4h*)&sB2[bLds[it]] = h2;
    }
    __syncthreads();
    // ---- prefetch next tile (drains during the WMMAs below) ----
    if (k0 + BK < DDIM) fetch(k0 + BK);
    // ---- fragments + WMMA ----
    FragAB a[2], b1[2], b2[2];
#pragma unroll
    for (int f = 0; f < 2; ++f) {
      int m = waveM * 32 + f * 16 + mRow;
      a[f].h[0] = *(const v8h*)&sA[m * LDK + khalf * 8];
      a[f].h[1] = *(const v8h*)&sA[m * LDK + khalf * 8 + 16];
      int n = waveN * 32 + f * 16 + mRow;
      b1[f].h[0] = *(const v8h*)&sB1[n * LDK + khalf * 8];
      b1[f].h[1] = *(const v8h*)&sB1[n * LDK + khalf * 8 + 16];
      b2[f].h[0] = *(const v8h*)&sB2[n * LDK + khalf * 8];
      b2[f].h[1] = *(const v8h*)&sB2[n * LDK + khalf * 8 + 16];
    }
#pragma unroll
    for (int fm = 0; fm < 2; ++fm)
#pragma unroll
      for (int fn = 0; fn < 2; ++fn) {
        accG[fm][fn] = __builtin_amdgcn_wmma_f32_16x16x32_f16(
            false, a[fm].v, false, b1[fn].v, (short)0, accG[fm][fn], false, false);
        accU[fm][fn] = __builtin_amdgcn_wmma_f32_16x16x32_f16(
            false, a[fm].v, false, b2[fn].v, (short)0, accU[fm][fn], false, false);
      }
  }
  // ---- SwiGLU epilogue: hidden = silu(g) * u, store f16 ----
#pragma unroll
  for (int fm = 0; fm < 2; ++fm)
#pragma unroll
    for (int fn = 0; fn < 2; ++fn)
#pragma unroll
      for (int r = 0; r < 8; ++r) {
        int mLocal = waveM * 32 + fm * 16 + r + khalf * 8;
        int gr = rowBase + mLocal;
        if (gr < count) {
          float g = accG[fm][fn][r];
          float u = accU[fm][fn][r];
          float h = (g / (1.f + __expf(-g))) * u;
          int col = hBase + waveN * 32 + fn * 16 + mRow;
          hidden[(size_t)(aOff + gr) * HDIM + col] = (_Float16)h;
        }
      }
}

// =====================================================================
// Kernel 4: out += prob * (hidden @ W3e^T)   (K = H = 2048)
// A tile is f16 already -> pure copy -> GLOBAL_LOAD_ASYNC_TO_LDS_B128.
// =====================================================================
__global__ __launch_bounds__(256)
void moe_expert_ffn2(const _Float16* __restrict__ hidden,
                     const float* __restrict__ W3,
                     const int* __restrict__ counts, const int* __restrict__ offsets,
                     const int* __restrict__ assign_token,
                     const float* __restrict__ assign_prob,
                     float* __restrict__ out) {
  const int e = blockIdx.z;
  const int count = counts[e];
  const int rowBase = blockIdx.y * BM;
  if (rowBase >= count) return;
  const int dBase = blockIdx.x * BN;
  const int aOff  = offsets[e];

  __shared__ _Float16 sA[BM * LDK];
  __shared__ _Float16 sB[BN * LDK];

  const int tid   = threadIdx.x;
  const int lane  = tid & 31;
  const int wave  = tid >> 5;
  const int waveM = wave >> 1;
  const int waveN = wave & 1;
  const int mRow  = lane & 15;
  const int khalf = lane >> 4;

  const float* W3e = W3 + (size_t)e * DDIM * HDIM;

  // ---- loop-invariant A-tile async-copy setup + LDS pre-zero ----
  // Invalid rows (gr >= count) never issue the async copy; their LDS slots
  // are zeroed once here and stay zero for the whole K loop.
  unsigned aLdsOff[2]; const char* aGsrc[2];
#pragma unroll
  for (int it = 0; it < 2; ++it) {
    int slot = tid + it * 256;               // 512 sixteen-byte slots (128x32 f16)
    int r = slot >> 2, c8 = slot & 3;
    int gr = rowBase + r;
    int idx = r * LDK + c8 * 8;
    *(v8h*)&sA[idx] = v8h_zero();
    aLdsOff[it] = (unsigned)(size_t)(lds_h*)&sA[idx];
    aGsrc[it] = (gr < count)
        ? (const char*)(hidden + (size_t)(aOff + gr) * HDIM + c8 * 8)
        : nullptr;
  }
  const float* bSrc[2]; int bLds[2];
#pragma unroll
  for (int it = 0; it < 2; ++it) {
    int slot = tid + it * 256;               // 512 float4-slots (64x32)
    int r = slot >> 3, c4 = slot & 7;
    bSrc[it] = W3e + (size_t)(dBase + r) * HDIM + c4 * 4;
    bLds[it] = r * LDK + c4 * 4;
  }

  v4f pB[2];
  auto fetchB = [&](int k0) {
#pragma unroll
    for (int it = 0; it < 2; ++it) pB[it] = *(const v4f*)(bSrc[it] + k0);
  };

  v8f acc[2][2];
#pragma unroll
  for (int i = 0; i < 2; ++i)
#pragma unroll
    for (int j = 0; j < 2; ++j) acc[i][j] = v8f_zero();

  fetchB(0);
  for (int k0 = 0; k0 < HDIM; k0 += BK) {
    __syncthreads();                         // LDS free from previous iteration
    // ---- A tile: async global->LDS copies (no VGPR round-trip) ----
#pragma unroll
    for (int it = 0; it < 2; ++it)
      if (aGsrc[it]) async_load_lds_b128(aLdsOff[it], aGsrc[it] + (size_t)k0 * 2);
    // ---- B tile: stage prefetched registers (fp32 -> f16) ----
#pragma unroll
    for (int it = 0; it < 2; ++it) {
      v4h h;
#pragma unroll
      for (int q = 0; q < 4; ++q) h[q] = (_Float16)pB[it][q];
      *(v4h*)&sB[bLds[it]] = h;
    }
    wait_asynccnt0();                        // our async copies landed in LDS
    __syncthreads();
    if (k0 + BK < HDIM) fetchB(k0 + BK);     // overlap with WMMAs
    FragAB a[2], b[2];
#pragma unroll
    for (int f = 0; f < 2; ++f) {
      int m = waveM * 32 + f * 16 + mRow;
      a[f].h[0] = *(const v8h*)&sA[m * LDK + khalf * 8];
      a[f].h[1] = *(const v8h*)&sA[m * LDK + khalf * 8 + 16];
      int n = waveN * 32 + f * 16 + mRow;
      b[f].h[0] = *(const v8h*)&sB[n * LDK + khalf * 8];
      b[f].h[1] = *(const v8h*)&sB[n * LDK + khalf * 8 + 16];
    }
#pragma unroll
    for (int fm = 0; fm < 2; ++fm)
#pragma unroll
      for (int fn = 0; fn < 2; ++fn)
        acc[fm][fn] = __builtin_amdgcn_wmma_f32_16x16x32_f16(
            false, a[fm].v, false, b[fn].v, (short)0, acc[fm][fn], false, false);
  }
  // ---- weighted atomic accumulate (exactly 2 contributions/element) ----
#pragma unroll
  for (int fm = 0; fm < 2; ++fm)
#pragma unroll
    for (int fn = 0; fn < 2; ++fn)
#pragma unroll
      for (int r = 0; r < 8; ++r) {
        int mLocal = waveM * 32 + fm * 16 + r + khalf * 8;
        int gr = rowBase + mLocal;
        if (gr < count) {
          int aIdx = aOff + gr;
          int tok  = assign_token[aIdx];
          float p  = assign_prob[aIdx];
          int col  = dBase + waveN * 32 + fn * 16 + mRow;
          atomicAdd(&out[(size_t)tok * DDIM + col], acc[fm][fn][r] * p);
        }
      }
}

// =====================================================================
// launch
// =====================================================================
extern "C" void kernel_launch(void* const* d_in, const int* in_sizes, int n_in,
                              void* d_out, int out_size, void* d_ws, size_t ws_size,
                              hipStream_t stream) {
  (void)in_sizes; (void)n_in; (void)out_size; (void)ws_size;
  const float* x  = (const float*)d_in[0];
  const float* Wg = (const float*)d_in[1];
  const float* W1 = (const float*)d_in[2];
  const float* W2 = (const float*)d_in[3];
  const float* W3 = (const float*)d_in[4];
  float* out = (float*)d_out;

  char* ws = (char*)d_ws;
  int*      counts       = (int*)(ws);
  int*      offsets      = (int*)(ws + 64);
  int*      top_idx      = (int*)(ws + 128);
  float*    top_prob     = (float*)(ws + 128 + 1 * 65536);
  int*      assign_token = (int*)(ws + 128 + 2 * 65536);
  float*    assign_prob  = (float*)(ws + 128 + 3 * 65536);
  _Float16* hidden       = (_Float16*)(ws + (1u << 20));   // NASSIGN*HDIM f16 = 64 MB

  moe_zero_out<<<(T_TOK * DDIM / 4 + 255) / 256, 256, 0, stream>>>(out, T_TOK * DDIM / 4);
  moe_gate_topk<<<T_TOK / 8, 256, 0, stream>>>(x, Wg, top_idx, top_prob);
  moe_build_lists<<<1, 64, 0, stream>>>(top_idx, top_prob, counts, offsets,
                                        assign_token, assign_prob);
  dim3 g1(HDIM / BN, NASSIGN / BM, NEXP);
  moe_expert_ffn1<<<g1, 256, 0, stream>>>(x, W1, W2, counts, offsets,
                                          assign_token, hidden);
  dim3 g2(DDIM / BN, NASSIGN / BM, NEXP);
  moe_expert_ffn2<<<g2, 256, 0, stream>>>(hidden, W3, counts, offsets,
                                          assign_token, assign_prob, out);
}
